// EcoAttention_89266600280757
// MI455X (gfx1250) — matrix operand order repaired and checked by reading, not compiled, not run on hardware
//
#include <hip/hip_runtime.h>

typedef __attribute__((ext_vector_type(16))) _Float16 v16h;
typedef __attribute__((ext_vector_type(8)))  _Float16 v8h;
typedef __attribute__((ext_vector_type(8)))  float    v8f;

#define SEQ    4096
#define DIM    128
#define BLK    64
#define NB     (SEQ / BLK)      // 64 key/query blocks
#define NCHUNK 8                // j-blocks split across 8 workgroups per query block
#define JPER   (NB / NCHUNK)    // 8 key blocks per workgroup

union AF { v16h v; v8h h[2]; };

__device__ __forceinline__ float wave_max32(float x) {
#pragma unroll
  for (int o = 16; o > 0; o >>= 1) x = fmaxf(x, __shfl_xor(x, o, 32));
  return x;
}
__device__ __forceinline__ float wave_sum32(float x) {
#pragma unroll
  for (int o = 16; o > 0; o >>= 1) x += __shfl_xor(x, o, 32);
  return x;
}

// ---------------------------------------------------------------------------
// prep: f32 -> f16 row-major copies of Q and K; f16 transpose of V (vT[d][s]).
// Puts all GEMM operands in a layout where WMMA fragments are contiguous
// 16B/32B loads per lane.
// ---------------------------------------------------------------------------
__global__ __launch_bounds__(256) void eco_prep(const float* __restrict__ q,
                                                const float* __restrict__ k,
                                                const float* __restrict__ v,
                                                _Float16* __restrict__ qh,
                                                _Float16* __restrict__ kh,
                                                _Float16* __restrict__ vT) {
  int idx = blockIdx.x * blockDim.x + threadIdx.x;
  if (idx >= SEQ * DIM) return;
  qh[idx] = (_Float16)q[idx];
  kh[idx] = (_Float16)k[idx];
  int col = idx / SEQ;   // vT layout: [DIM][SEQ]
  int row = idx % SEQ;
  vT[idx] = (_Float16)v[row * DIM + col];
}

// ---------------------------------------------------------------------------
// main attention kernel: one workgroup = (query block ib) x (chunk of 8 key
// blocks). 4 waves; wave w owns query rows [ib*64 + w*16, +16). All LDS tiles
// are wave-private -> zero barriers.
// ---------------------------------------------------------------------------
__global__ __launch_bounds__(128) void eco_attn(const _Float16* __restrict__ qh,
                                                const _Float16* __restrict__ kh,
                                                const _Float16* __restrict__ vT,
                                                float* __restrict__ pout) {
  __shared__ float    s_sc[4][16][64];   // 16 KB: raw scores per wave strip
  __shared__ _Float16 s_p [4][16][64];   //  8 KB: masked/normalized weights

  const int ib    = blockIdx.x;          // query block
  const int chunk = blockIdx.y;          // key-block chunk
  const int w     = threadIdx.x >> 5;    // wave id = strip id
  const int lane  = threadIdx.x & 31;
  const int g     = lane >> 4;           // lane group (0/1)
  const int ln    = lane & 15;

  // --- resident Q A-fragments for this strip (4 K-steps of 32) ---
  // A operand, contiguous k order chosen here: element e -> K = g*16 + e
  v16h qa[4];
  {
    const _Float16* qrow = qh + (ib * 64 + w * 16 + ln) * DIM;
#pragma unroll
    for (int s = 0; s < 4; ++s) {
      AF t;
      t.h[0] = *(const v8h*)(qrow + s * 32 + g * 16);        // e=0..7
      t.h[1] = *(const v8h*)(qrow + s * 32 + g * 16 + 8);   // e=8..15
      qa[s] = t.v;
    }
  }

  v8f oacc[8];
#pragma unroll
  for (int n = 0; n < 8; ++n) oacc[n] = (v8f){};

  for (int jj = 0; jj < JPER; ++jj) {
    const int j = chunk * JPER + jj;

    // ---- scores: S_strip(16x64) = Q_strip(16x128) @ K_j^T ----
    // B operand, same contiguous k order as A: element e -> K = g*16 + e (contiguous in d)
#pragma unroll
    for (int n = 0; n < 4; ++n) {
      v8f c = (v8f){};
      const _Float16* krow = kh + (j * 64 + n * 16 + ln) * DIM;
#pragma unroll
      for (int s = 0; s < 4; ++s) {
        v16h b = *(const v16h*)(krow + s * 32 + g * 16);
        c = __builtin_amdgcn_wmma_f32_16x16x32_f16(false, qa[s], false, b,
                                                   (short)0, c, false, false);
      }
      // C-layout: VGPR r holds row M = g*8 + r, col N = n*16 + ln
#pragma unroll
      for (int r = 0; r < 8; ++r)
        s_sc[w][g * 8 + r][n * 16 + ln] = c[r];
    }

    // ---- per-row top-p via bitonic sort + prefix scan; one wave per row ----
    // Row of 64 exp-values lives 2-per-lane: index c0 = lane, c1 = lane+32.
    // We need t = #{descending-sorted inclusive cumsums < 0.95*sum}; the mask
    // then keeps ORIGINAL indices c < t (faithful to the reference).
    for (int row = 0; row < 16; ++row) {
      float s0 = s_sc[w][row][lane];
      float s1 = s_sc[w][row][lane + 32];
      float m  = wave_max32(fmaxf(s0, s1));
      float e0 = __expf(s0 - m);
      float e1 = __expf(s1 - m);

      // bitonic sort (descending) of a copy; 64 elems, virtual idx i = lane + 32*h
      float x0 = e0, x1 = e1;
#pragma unroll
      for (int bk = 2; bk <= 64; bk <<= 1) {
#pragma unroll
        for (int bj = bk >> 1; bj > 0; bj >>= 1) {
          if (bj == 32) {                      // partner is the other register
            float hi = fmaxf(x0, x1), lo = fminf(x0, x1);
            x0 = hi; x1 = lo;                  // (i&64)==0 -> lower idx keeps max
          } else {
            float p0 = __shfl_xor(x0, bj, 32);
            float p1 = __shfl_xor(x1, bj, 32);
            bool b0 = (((lane     ) & bk) == 0) == ((lane & bj) == 0);
            bool b1 = (((lane + 32) & bk) == 0) == ((lane & bj) == 0);
            x0 = b0 ? fmaxf(x0, p0) : fminf(x0, p0);
            x1 = b1 ? fmaxf(x1, p1) : fminf(x1, p1);
          }
        }
      }

      // inclusive prefix sums of the sorted (descending) sequence
#pragma unroll
      for (int d = 1; d < 32; d <<= 1) {
        float t0 = __shfl_up(x0, d, 32);
        float t1 = __shfl_up(x1, d, 32);
        if (lane >= d) { x0 += t0; x1 += t1; }
      }
      float tot0  = __shfl(x0, 31, 32);        // sum of top 32
      float tot   = tot0 + __shfl(x1, 31, 32); // row sum S (from the scan, free)
      float theta = 0.95f * tot;
      int t = __popcll(__ballot(x0 < theta)) +
              __popcll(__ballot((x1 + tot0) < theta));

      // keep original indices c < t, renormalize
      float k0 = (lane      < t) ? e0 : 0.0f;
      float k1 = (lane + 32 < t) ? e1 : 0.0f;
      float inv = 1.0f / (wave_sum32(k0 + k1) + 1e-8f);
      s_p[w][row][lane]      = (_Float16)(k0 * inv);
      s_p[w][row][lane + 32] = (_Float16)(k1 * inv);
    }

    // ---- PV: O_strip(16x128) += P(16x64) @ V_j(64x128) ----
#pragma unroll
    for (int s = 0; s < 2; ++s) {
      AF pa;
      const _Float16* prow = &s_p[w][ln][0];
      pa.h[0] = *(const v8h*)(prow + s * 32 + g * 16);
      pa.h[1] = *(const v8h*)(prow + s * 32 + g * 16 + 8);
#pragma unroll
      for (int n = 0; n < 8; ++n) {
        v16h b = *(const v16h*)(vT + (n * 16 + ln) * SEQ + j * 64 + s * 32 + g * 16);
        oacc[n] = __builtin_amdgcn_wmma_f32_16x16x32_f16(false, pa.v, false, b,
                                                         (short)0, oacc[n], false, false);
      }
    }
  }

  // partial output for this chunk (exclusive region -> deterministic)
  float* outp = pout + (size_t)chunk * SEQ * DIM;
#pragma unroll
  for (int n = 0; n < 8; ++n) {
    const int col = n * 16 + ln;
#pragma unroll
    for (int r = 0; r < 8; ++r) {
      const int row = ib * 64 + w * 16 + g * 8 + r;
      outp[row * DIM + col] = oacc[n][r];
    }
  }
}

// ---------------------------------------------------------------------------
// deterministic fixed-order reduction of the NCHUNK partials
// ---------------------------------------------------------------------------
__global__ __launch_bounds__(256) void eco_reduce(const float* __restrict__ pout,
                                                  float* __restrict__ out) {
  int idx = blockIdx.x * blockDim.x + threadIdx.x;
  if (idx >= SEQ * DIM) return;
  float acc = 0.0f;
#pragma unroll
  for (int c = 0; c < NCHUNK; ++c) acc += pout[(size_t)c * SEQ * DIM + idx];
  out[idx] = acc;
}

extern "C" void kernel_launch(void* const* d_in, const int* in_sizes, int n_in,
                              void* d_out, int out_size, void* d_ws, size_t ws_size,
                              hipStream_t stream) {
  const float* q = (const float*)d_in[0];
  const float* k = (const float*)d_in[1];
  const float* v = (const float*)d_in[2];
  float* out = (float*)d_out;

  // workspace layout: 3 MB f16 operands + 16 MB f32 chunk partials (~19 MB)
  _Float16* qh = (_Float16*)d_ws;
  _Float16* kh = qh + SEQ * DIM;
  _Float16* vT = kh + SEQ * DIM;
  float* pout  = (float*)(vT + SEQ * DIM);

  const int n = SEQ * DIM;
  eco_prep<<<(n + 255) / 256, 256, 0, stream>>>(q, k, v, qh, kh, vT);
  eco_attn<<<dim3(NB, NCHUNK), 128, 0, stream>>>(qh, kh, vT, pout);
  eco_reduce<<<(n + 255) / 256, 256, 0, stream>>>(pout, out);
}
